// CrossAttentionPool_2594160246937
// MI455X (gfx1250) — matrix operand order repaired and checked by reading, not compiled, not run on hardware
//
#include <hip/hip_runtime.h>
#include <math.h>

// CDNA5 / gfx1250 fused cross-attention pool using V_WMMA_F32_16X16X32_BF16.
// Memory-bound (~120MB HBM traffic vs ~30 GFLOP): one pass over `lines`,
// K precomputed once into workspace (bf16), intermediates in LDS.
// Round 2: native bf16 converts (v_cvt_pk_bf16_f32) + b128 fragment loads.

typedef __attribute__((ext_vector_type(16))) __bf16 v16bf;
typedef __attribute__((ext_vector_type(8)))  float  v8f;

constexpr int NB = 32;   // batch
constexpr int NL = 512;  // lines per batch
constexpr int NV = 128;  // videos per batch
constexpr int ND = 768;  // feature dim
constexpr int LT = 16;   // L-rows per workgroup in fused kernel

__device__ __forceinline__ __bf16 f2bf(float f) { return (__bf16)f; }

// ---- WMMA fragment loaders (wave32, 16x16x32 bf16) ----------------------
// A (16x32, MxK): lanes 0-15 row=lane, K=0..15; lanes 16-31 K=16..31 (contiguous)
__device__ __forceinline__ v16bf a_frag_f32(const float* src, int ld, int lane) {
  const float* p = src + (lane & 15) * ld + ((lane >> 4) << 4);
  float4 x0 = *(const float4*)(p);       // K+0..3
  float4 x1 = *(const float4*)(p + 4);   // K+4..7
  float4 x2 = *(const float4*)(p + 8);   // K+8..11
  float4 x3 = *(const float4*)(p + 12);  // K+12..15
  v16bf a;
  a[0] = f2bf(x0.x); a[1] = f2bf(x0.y); a[2] = f2bf(x0.z); a[3] = f2bf(x0.w);
  a[4] = f2bf(x1.x); a[5] = f2bf(x1.y); a[6] = f2bf(x1.z); a[7] = f2bf(x1.w);
  a[8]  = f2bf(x2.x); a[9]  = f2bf(x2.y); a[10] = f2bf(x2.z); a[11] = f2bf(x2.w);
  a[12] = f2bf(x3.x); a[13] = f2bf(x3.y); a[14] = f2bf(x3.z); a[15] = f2bf(x3.w);
  return a;
}
__device__ __forceinline__ v16bf a_frag_bf16(const __bf16* src, int ld, int lane) {
  const __bf16* p = src + (lane & 15) * ld + ((lane >> 4) << 4);
  v16bf a;
  __builtin_memcpy(&a, p, 32);                        // elements 0..15 (K+0..15)
  return a;
}
// B (32x16, KxN) where B[k][n] = W[n][k], W row-major, leading dim ld.
// lanes 0-15: n=lane, K=0..15 ; lanes 16-31: n=lane-16, K=16..31 (contiguous)
__device__ __forceinline__ v16bf bT_frag_f32(const float* src, int ld, int lane) {
  const float* p = src + (lane & 15) * ld + ((lane >> 4) << 4);
  float4 x0 = *(const float4*)(p);
  float4 x1 = *(const float4*)(p + 4);
  float4 x2 = *(const float4*)(p + 8);
  float4 x3 = *(const float4*)(p + 12);
  v16bf b;
  b[0] = f2bf(x0.x); b[1] = f2bf(x0.y); b[2]  = f2bf(x0.z); b[3]  = f2bf(x0.w);
  b[4] = f2bf(x1.x); b[5] = f2bf(x1.y); b[6]  = f2bf(x1.z); b[7]  = f2bf(x1.w);
  b[8] = f2bf(x2.x); b[9] = f2bf(x2.y); b[10] = f2bf(x2.z); b[11] = f2bf(x2.w);
  b[12] = f2bf(x3.x); b[13] = f2bf(x3.y); b[14] = f2bf(x3.z); b[15] = f2bf(x3.w);
  return b;
}
__device__ __forceinline__ v16bf bT_frag_bf16(const __bf16* src, int ld, int lane) {
  const __bf16* p = src + (lane & 15) * ld + ((lane >> 4) << 4);
  v16bf b;
  __builtin_memcpy(&b, p, 32);  // 16 contiguous bf16 -> b128 x2
  return b;
}
// B (32x16, KxN) where B[k][n] = M[k][n], M row-major, leading dim ld (strided)
__device__ __forceinline__ v16bf b_frag_f32(const float* src, int ld, int lane) {
  const float* p = src + ((lane >> 4) << 4) * ld + (lane & 15);
  v16bf b;
#pragma unroll
  for (int j = 0; j < 16; ++j) b[j] = f2bf(p[j * ld]);
  return b;
}

__device__ __forceinline__ v8f wmma_bf16(v16bf a, v16bf b, v8f c) {
  // (neg_a, A, neg_b, B, c_mod, C, reuse_a, reuse_b)
  return __builtin_amdgcn_wmma_f32_16x16x32_bf16(false, a, false, b, (short)0, c,
                                                 false, false);
}

// ================= Kernel 1: kmat[b] = videos[b] @ w_k^T (bf16) ==========
__global__ __launch_bounds__(256) void kproj_kernel(
    const float* __restrict__ videos, const float* __restrict__ w_k,
    __bf16* __restrict__ kmat) {
  const int b    = blockIdx.x;
  const int lane = threadIdx.x & 31;
  const int wave = threadIdx.x >> 5;

  const float* vb = videos + (size_t)b * NV * ND;
  __bf16* kb = kmat + (size_t)b * NV * ND;

  const int ntiles = (NV / 16) * (ND / 16);  // 8 * 48 = 384
  for (int t = wave; t < ntiles; t += 8) {
    const int v0 = (t / (ND / 16)) * 16;
    const int e0 = (t % (ND / 16)) * 16;
    v8f acc = {};
#pragma unroll 4
    for (int kc = 0; kc < ND; kc += 32) {
      v16bf a  = a_frag_f32(vb + v0 * ND + kc, ND, lane);
      v16bf bm = bT_frag_f32(w_k + e0 * ND + kc, ND, lane);  // B[k][n]=w_k[e0+n][kc+k]
      acc = wmma_bf16(a, bm, acc);
    }
    const int row = v0 + ((lane >> 4) << 3);
    const int col = e0 + (lane & 15);
#pragma unroll
    for (int r = 0; r < 8; ++r) kb[(size_t)(row + r) * ND + col] = f2bf(acc[r]);
  }
}

// ====== Kernel 2: fused Q-proj -> scores -> softmax -> weighted sum ======
__global__ __launch_bounds__(256) void attn_kernel(
    const float* __restrict__ lines, const float* __restrict__ videos,
    const int* __restrict__ vmask, const float* __restrict__ w_q,
    const __bf16* __restrict__ kmat, float* __restrict__ out) {
  const int b    = blockIdx.y;
  const int l0   = blockIdx.x * LT;
  const int lane = threadIdx.x & 31;
  const int wave = threadIdx.x >> 5;

  __shared__ __bf16 q_lds[LT * ND];  // 24 KB  (Q tile, bf16)
  __shared__ float  s_lds[LT * NV];  //  8 KB  (scores, f32)
  __shared__ __bf16 w_lds[LT * NV];  //  4 KB  (softmax weights, bf16)

  const float* lb = lines + ((size_t)b * NL + l0) * ND;

  // ---- Phase A: Q = lines_tile @ w_q^T  -> q_lds ----
  for (int t = wave; t < ND / 16; t += 8) {
    const int e0 = t * 16;
    v8f acc = {};
#pragma unroll 4
    for (int kc = 0; kc < ND; kc += 32) {
      v16bf a  = a_frag_f32(lb + kc, ND, lane);
      v16bf bm = bT_frag_f32(w_q + e0 * ND + kc, ND, lane);  // B[k][n]=w_q[e0+n][kc+k]
      acc = wmma_bf16(a, bm, acc);
    }
    const int row = (lane >> 4) << 3;
    const int col = e0 + (lane & 15);
#pragma unroll
    for (int r = 0; r < 8; ++r) q_lds[(row + r) * ND + col] = f2bf(acc[r]);
  }
  __syncthreads();

  // ---- Phase B: scores = (Q @ K^T) * scale, masked -> s_lds ----
  {
    const __bf16* kb = kmat + (size_t)b * NV * ND;
    const int v0 = wave * 16;  // 8 waves cover V = 128
    v8f acc = {};
#pragma unroll 4
    for (int kc = 0; kc < ND; kc += 32) {
      v16bf a  = a_frag_bf16(q_lds + kc, ND, lane);
      v16bf bm = bT_frag_bf16(kb + v0 * ND + kc, ND, lane);  // B[k][n]=kmat[v0+n][kc+k]
      acc = wmma_bf16(a, bm, acc);
    }
    const float scale = 0.03608439182435161f;  // 768^-0.5
    const int row = (lane >> 4) << 3;
    const int col = v0 + (lane & 15);
    const int m = vmask[b * NV + col];
#pragma unroll
    for (int r = 0; r < 8; ++r) {
      float s = acc[r] * scale;
      s_lds[(row + r) * NV + col] = (m == 0) ? -1e9f : s;
    }
  }
  __syncthreads();

  // ---- Softmax over V per row -> w_lds (bf16) ----
  if (threadIdx.x < LT) {
    const int r = threadIdx.x;
    float mx = -3.4e38f;
    for (int v = 0; v < NV; ++v) mx = fmaxf(mx, s_lds[r * NV + v]);
    float sum = 0.f;
    for (int v = 0; v < NV; ++v) sum += __expf(s_lds[r * NV + v] - mx);
    const float inv = 1.0f / sum;
    for (int v = 0; v < NV; ++v)
      w_lds[r * NV + v] = f2bf(__expf(s_lds[r * NV + v] - mx) * inv);
  }
  __syncthreads();

  // ---- Phase C: out = weights @ videos ----
  const float* vb = videos + (size_t)b * NV * ND;
  float* ob = out + ((size_t)b * NL + l0) * ND;
  for (int t = wave; t < ND / 16; t += 8) {
    const int e0 = t * 16;
    v8f acc = {};
#pragma unroll
    for (int kc = 0; kc < NV; kc += 32) {
      v16bf a  = a_frag_bf16(w_lds + kc, NV, lane);
      v16bf bm = b_frag_f32(vb + (size_t)kc * ND + e0, ND, lane);  // B[k][n]=videos[kc+k][e0+n]
      acc = wmma_bf16(a, bm, acc);
    }
    const int row = (lane >> 4) << 3;
    const int col = e0 + (lane & 15);
#pragma unroll
    for (int r = 0; r < 8; ++r) ob[(size_t)(row + r) * ND + col] = acc[r];
  }
}

extern "C" void kernel_launch(void* const* d_in, const int* in_sizes, int n_in,
                              void* d_out, int out_size, void* d_ws, size_t ws_size,
                              hipStream_t stream) {
  (void)in_sizes; (void)n_in; (void)out_size; (void)ws_size;
  const float* lines  = (const float*)d_in[0];  // (32,512,768) f32
  const float* videos = (const float*)d_in[1];  // (32,128,768) f32
  const int*   vmask  = (const int*)d_in[2];    // (32,128)     i32
  const float* w_q    = (const float*)d_in[3];  // (768,768)    f32
  const float* w_k    = (const float*)d_in[4];  // (768,768)    f32
  float* out = (float*)d_out;                   // (32,512,768) f32
  __bf16* kmat = (__bf16*)d_ws;                 // (32,128,768) bf16 = 6 MB scratch

  kproj_kernel<<<dim3(NB), 256, 0, stream>>>(videos, w_k, kmat);
  attn_kernel<<<dim3(NL / LT, NB), 256, 0, stream>>>(lines, videos, vmask, w_q,
                                                     kmat, out);
}
